// Seq2Seq_31473520345563
// MI455X (gfx1250) — matrix operand order repaired and checked by reading, not compiled, not run on hardware
//
#include <hip/hip_runtime.h>
#include <hip/hip_bf16.h>

// ---------------------------------------------------------------------------
// Seq2Seq LSTM enc/dec on MI455X (gfx1250, wave32, WMMA + async-LDS staging).
//   * bf16 weights converted once -> L2-resident (~58MB of 192MB L2).
//   * GEMM: each wave owns one 16-col N-tile and all 4 M-tiles (M=64), so
//     every weight fragment feeds 4 v_wmma_f32_16x16x32_bf16 ops and is read
//     exactly once device-wide.
//   * Activation panel A[64,K] staged into LDS chunks via
//     global_load_async_to_lds_b128 (double-buffered, s_wait_asynccnt).
// ---------------------------------------------------------------------------

typedef __attribute__((ext_vector_type(16))) __bf16 v16bf;
typedef __attribute__((ext_vector_type(8)))  __bf16 v8bf;
typedef __attribute__((ext_vector_type(8)))  float  v8f;

#define BATCH   64
#define EDIM    512
#define HDIM    1024
#define H4      4096
#define SRCLEN  128
#define TGTLEN  128

// --- LDS staging geometry ---
#define KCHUNK   128                  // K-values per staged chunk
#define ROWPAD   8                    // 16B pad -> conflict-free fragment reads
#define ROWLEN   (KCHUNK + ROWPAD)    // 136 bf16 per row
#define BUF_ELEMS (64 * ROWLEN)       // 8704 bf16 = 17408 B per buffer
#define GEMM_THREADS 128              // 4 waves per block
#define CHUNK_UNITS  (64 * KCHUNK / 8)            // 16B units per chunk (1024)

__device__ __forceinline__ v16bf cat8(v8bf lo, v8bf hi) {
  return __builtin_shufflevector(lo, hi, 0,1,2,3,4,5,6,7,8,9,10,11,12,13,14,15);
}

__device__ __forceinline__ float sigmoidf_(float x) {
  return 1.0f / (1.0f + __expf(-x));
}

// gfx1250 async copy: 16B per lane, global -> LDS, tracked by ASYNCcnt.
__device__ __forceinline__ void async_ld_b128(unsigned lds_off, const __bf16* g) {
  asm volatile("global_load_async_to_lds_b128 %0, %1, off"
               :: "v"(lds_off), "v"(g) : "memory");
}

// ---------------- utility kernels ----------------

__global__ void k_f32_to_bf16(const float* __restrict__ s, __bf16* __restrict__ d, int n) {
  int i = blockIdx.x * blockDim.x + threadIdx.x;
  if (i < n) d[i] = (__bf16)s[i];
}

__global__ void k_zero_f32(float* p, int n) {
  int i = blockIdx.x * blockDim.x + threadIdx.x;
  if (i < n) p[i] = 0.0f;
}

__global__ void k_zero_bf16(__bf16* p, int n) {
  int i = blockIdx.x * blockDim.x + threadIdx.x;
  if (i < n) p[i] = (__bf16)0.0f;
}

__global__ void k_zero_i32(int* p, int n) {
  int i = blockIdx.x * blockDim.x + threadIdx.x;
  if (i < n) p[i] = 0;
}

// Gather embedding rows (f32) for this timestep, emit bf16 activations [B, E].
__global__ void k_embed(const int* __restrict__ ids, int stride, int off,
                        const float* __restrict__ emb, int edim,
                        __bf16* __restrict__ xe) {
  int i = blockIdx.x * blockDim.x + threadIdx.x;       // B*edim threads
  int b = i / edim, e = i - b * edim;
  int id = ids[b * stride + off];
  xe[i] = (__bf16)emb[(long)id * edim + e];
}

// ---------------- WMMA GEMM with async-LDS A staging ----------------
// Z[64, N] = A0[64,K0] @ W0[N,K0]^T + A1[64,K1] @ W1[N,K1]^T
// Block: 128 threads = 4 waves; wave w -> N-tile n0 = blk*64 + w*16,
// all four 16-row M-tiles accumulated per wave.  grid.x = N/64.

__device__ __forceinline__ v16bf lds_a_frag(const __bf16* __restrict__ buf,
                                            int m0, int lane, int klocal) {
  // A-matrix 16x32 bf16 layout: lane<16 -> K {0..7,16..23}, lane>=16 -> {8..15,24..31}.
  int row = m0 + (lane & 15);
  int kb  = klocal + ((lane >> 4) << 3);
  const __bf16* p = buf + row * ROWLEN + kb;
  v8bf lo = *(const v8bf*)(p);
  v8bf hi = *(const v8bf*)(p + 16);
  return cat8(lo, hi);
}

__device__ __forceinline__ v16bf load_b_frag(const __bf16* __restrict__ W, int K,
                                             int n0, int lane, int k0) {
  // B-matrix 32x16: lane<16 holds K {0..7,16..23}, lane>=16 -> {8..15,24..31} of column n0+lane (B[k][n]=W[n][k]).
  int row = n0 + (lane & 15);
  int kb  = k0 + ((lane >> 4) << 3);
  const __bf16* p = W + (long)row * K + kb;
  v8bf lo = *(const v8bf*)(p);
  v8bf hi = *(const v8bf*)(p + 16);
  return cat8(lo, hi);
}

__device__ __forceinline__ void stage_chunk(__bf16* __restrict__ buf,
                                            const __bf16* __restrict__ A,
                                            int K, int kc, int tid) {
  // 64 rows x KCHUNK cols in 16B units; 1024 units / 128 threads = 8 async ops/wave.
#pragma unroll
  for (int u = tid; u < CHUNK_UNITS; u += GEMM_THREADS) {
    int row = u >> 4;              // 16 units per row
    int k8  = (u & 15) << 3;
    unsigned lofs = (unsigned)(size_t)(buf + row * ROWLEN + k8);
    async_ld_b128(lofs, A + (size_t)row * K + kc + k8);
  }
}

__global__ void __launch_bounds__(GEMM_THREADS)
k_gemm_wmma(const __bf16* __restrict__ A0, const __bf16* __restrict__ W0, int K0,
            const __bf16* __restrict__ A1, const __bf16* __restrict__ W1, int K1,
            float* __restrict__ Z, int N) {
  extern __shared__ __bf16 smem[];

  const int tid  = threadIdx.x;
  const int lane = tid & 31;
  const int wave = tid >> 5;                       // 0..3
  const int n0   = blockIdx.x * 64 + wave * 16;

  v8f acc0 = {}, acc1 = {}, acc2 = {}, acc3 = {};

  for (int p = 0; p < 2; ++p) {
    const __bf16* A = p ? A1 : A0;
    const __bf16* W = p ? W1 : W0;
    const int     K = p ? K1 : K0;
    if (K == 0) continue;
    const int nch = K / KCHUNK;

    stage_chunk(smem, A, K, 0, tid);
    for (int c = 0; c < nch; ++c) {
      if (c + 1 < nch) {
        stage_chunk(smem + (size_t)((c + 1) & 1) * BUF_ELEMS, A, K, (c + 1) * KCHUNK, tid);
        asm volatile("s_wait_asynccnt 8" ::: "memory");   // chunk c landed, c+1 in flight
      } else {
        asm volatile("s_wait_asynccnt 0" ::: "memory");
      }
      __syncthreads();

      const __bf16* cb = smem + (size_t)(c & 1) * BUF_ELEMS;
#pragma unroll
      for (int kl = 0; kl < KCHUNK; kl += 32) {
        const int kg = c * KCHUNK + kl;
        __builtin_prefetch(W + (long)(n0 + (lane & 15)) * K + kg + 64, 0, 0);
        v16bf b  = load_b_frag(W, K, n0, lane, kg);
        v16bf a0 = lds_a_frag(cb,  0, lane, kl);
        v16bf a1 = lds_a_frag(cb, 16, lane, kl);
        v16bf a2 = lds_a_frag(cb, 32, lane, kl);
        v16bf a3 = lds_a_frag(cb, 48, lane, kl);
        acc0 = __builtin_amdgcn_wmma_f32_16x16x32_bf16(false, a0, false, b, (short)0, acc0, false, false);
        acc1 = __builtin_amdgcn_wmma_f32_16x16x32_bf16(false, a1, false, b, (short)0, acc1, false, false);
        acc2 = __builtin_amdgcn_wmma_f32_16x16x32_bf16(false, a2, false, b, (short)0, acc2, false, false);
        acc3 = __builtin_amdgcn_wmma_f32_16x16x32_bf16(false, a3, false, b, (short)0, acc3, false, false);
      }
      __syncthreads();   // protect buffer reuse by the next stage
    }
  }

  // C/D f32 layout: VGPR r -> M = m0 + r (+8 for lanes 16..31), N = lane&15.
  const int col = n0 + (lane & 15);
  const int rb  = (lane >> 4) << 3;
#pragma unroll
  for (int r = 0; r < 8; ++r) {
    Z[(long)(rb + r)      * N + col] = acc0[r];
    Z[(long)(16 + rb + r) * N + col] = acc1[r];
    Z[(long)(32 + rb + r) * N + col] = acc2[r];
    Z[(long)(48 + rb + r) * N + col] = acc3[r];
  }
}

// ---------------- LSTM gate update ----------------

__global__ void k_lstm_update(const float* __restrict__ Z, const float* __restrict__ bias,
                              float* __restrict__ c, __bf16* __restrict__ h) {
  int idx = blockIdx.x * blockDim.x + threadIdx.x;   // B*H
  int b  = idx >> 10;
  int hh = idx & (HDIM - 1);
  const float* zr = Z + (long)b * H4;
  float gi = zr[hh]          + bias[hh];
  float gf = zr[HDIM   + hh] + bias[HDIM   + hh];
  float gg = zr[2*HDIM + hh] + bias[2*HDIM + hh];
  float go = zr[3*HDIM + hh] + bias[3*HDIM + hh];
  float cc = sigmoidf_(gf) * c[idx] + sigmoidf_(gi) * tanhf(gg);
  c[idx] = cc;
  h[idx] = (__bf16)(sigmoidf_(go) * tanhf(cc));
}

// ---------------- decoder output: log-softmax + argmax + NLL ----------------

__global__ void k_dec_out(const float* __restrict__ Z, const float* __restrict__ linb,
                          const int* __restrict__ tag, int t,
                          float* __restrict__ logp_out,
                          float* __restrict__ nll, int* __restrict__ deint) {
  const int b   = blockIdx.x;
  const int tid = threadIdx.x;
  const float* zr = Z + (long)b * HDIM;

  __shared__ float smax[256];
  __shared__ int   sidx[256];
  __shared__ float ssum[256];

  float vals[4];
  float lmax = -INFINITY; int limidx = 0;
#pragma unroll
  for (int j = 0; j < 4; ++j) {
    int k = tid + j * 256;
    float v = zr[k] + linb[k];
    vals[j] = v;
    if (v > lmax) { lmax = v; limidx = k; }
  }
  smax[tid] = lmax; sidx[tid] = limidx;
  __syncthreads();
  for (int s = 128; s > 0; s >>= 1) {
    if (tid < s) {
      float ov = smax[tid + s]; int oi = sidx[tid + s];
      if (ov > smax[tid] || (ov == smax[tid] && oi < sidx[tid])) { smax[tid] = ov; sidx[tid] = oi; }
    }
    __syncthreads();
  }
  const float rmax = smax[0];
  const int   ridx = sidx[0];
  __syncthreads();

  float lsum = 0.0f;
#pragma unroll
  for (int j = 0; j < 4; ++j) lsum += __expf(vals[j] - rmax);
  ssum[tid] = lsum;
  __syncthreads();
  for (int s = 128; s > 0; s >>= 1) {
    if (tid < s) ssum[tid] += ssum[tid + s];
    __syncthreads();
  }
  const float lse = __logf(ssum[0]);

#pragma unroll
  for (int j = 0; j < 4; ++j) {
    int k = tid + j * 256;
    logp_out[(long)b * HDIM + k] = vals[j] - rmax - lse;
  }
  if (tid == 0) {
    deint[b] = ridx;                                 // greedy feedback
    int tg = tag[b * TGTLEN + t];
    float tv = zr[tg] + linb[tg];
    nll[b] = (tg != 0) ? -(tv - rmax - lse) : 0.0f;  // PAD==0 masked
  }
}

__global__ void k_step_loss(const float* __restrict__ nll, const int* __restrict__ tag,
                            int t, float* __restrict__ loss) {
  __shared__ float s[64];
  __shared__ int   m[64];
  int tid = threadIdx.x;   // 64 threads
  s[tid] = nll[tid];
  m[tid] = (tag[tid * TGTLEN + t] != 0) ? 1 : 0;
  __syncthreads();
  for (int k = 32; k > 0; k >>= 1) {
    if (tid < k) { s[tid] += s[tid + k]; m[tid] += m[tid + k]; }
    __syncthreads();
  }
  if (tid == 0) {
    int denom = m[0] > 0 ? m[0] : 1;
    loss[0] += s[0] / (float)denom;                  // single writer, stream-ordered
  }
}

// ---------------------------------------------------------------------------

extern "C" void kernel_launch(void* const* d_in, const int* in_sizes, int n_in,
                              void* d_out, int out_size, void* d_ws, size_t ws_size,
                              hipStream_t stream) {
  (void)in_sizes; (void)n_in; (void)out_size; (void)ws_size;

  const int*   input_ids = (const int*)  d_in[0];
  const int*   tag_ids   = (const int*)  d_in[1];
  const float* enc_embed = (const float*)d_in[2];
  const float* enc_Wih0  = (const float*)d_in[3];
  const float* enc_Whh0  = (const float*)d_in[4];
  const float* enc_b0    = (const float*)d_in[5];
  const float* enc_Wih1  = (const float*)d_in[6];
  const float* enc_Whh1  = (const float*)d_in[7];
  const float* enc_b1    = (const float*)d_in[8];
  const float* dec_embed = (const float*)d_in[9];
  const float* dec_Wih0  = (const float*)d_in[10];
  const float* dec_Whh0  = (const float*)d_in[11];
  const float* dec_b0    = (const float*)d_in[12];
  const float* dec_Wih1  = (const float*)d_in[13];
  const float* dec_Whh1  = (const float*)d_in[14];
  const float* dec_b1    = (const float*)d_in[15];
  const float* dec_linW  = (const float*)d_in[16];
  const float* dec_linb  = (const float*)d_in[17];

  float* out  = (float*)d_out;
  float* loss = out + (long)TGTLEN * BATCH * HDIM;

  // ---- workspace carve-out (bf16 weights stay L2-resident: ~58MB) ----
  char* ws = (char*)d_ws;
  size_t off = 0;
  auto carve = [&](size_t bytes) -> char* {
    char* p = ws + off;
    off = (off + bytes + 255) & ~(size_t)255;
    return p;
  };
  const size_t nWih0 = (size_t)H4 * EDIM, nWhh = (size_t)H4 * HDIM, nLin = (size_t)HDIM * HDIM;

  __bf16* eWih0 = (__bf16*)carve(nWih0 * 2);
  __bf16* eWhh0 = (__bf16*)carve(nWhh  * 2);
  __bf16* eWih1 = (__bf16*)carve(nWhh  * 2);
  __bf16* eWhh1 = (__bf16*)carve(nWhh  * 2);
  __bf16* dWih0 = (__bf16*)carve(nWih0 * 2);
  __bf16* dWhh0 = (__bf16*)carve(nWhh  * 2);
  __bf16* dWih1 = (__bf16*)carve(nWhh  * 2);
  __bf16* dWhh1 = (__bf16*)carve(nWhh  * 2);
  __bf16* linW  = (__bf16*)carve(nLin  * 2);

  __bf16* xe  = (__bf16*)carve((size_t)BATCH * EDIM * 2);
  __bf16* h0  = (__bf16*)carve((size_t)BATCH * HDIM * 2);
  __bf16* h1  = (__bf16*)carve((size_t)BATCH * HDIM * 2);
  float*  c0  = (float*) carve((size_t)BATCH * HDIM * 4);
  float*  c1  = (float*) carve((size_t)BATCH * HDIM * 4);
  float*  z   = (float*) carve((size_t)BATCH * H4   * 4);
  float*  zlin= (float*) carve((size_t)BATCH * HDIM * 4);
  float*  nll = (float*) carve((size_t)BATCH * 4);
  int*    dei = (int*)   carve((size_t)BATCH * 4);

  const int T256 = 256;
  auto cvt = [&](const float* s, __bf16* d, size_t n) {
    k_f32_to_bf16<<<(unsigned)((n + T256 - 1) / T256), T256, 0, stream>>>(s, d, (int)n);
  };

  // ---- one-time (per launch) weight conversion f32 -> bf16 ----
  cvt(enc_Wih0, eWih0, nWih0);  cvt(enc_Whh0, eWhh0, nWhh);
  cvt(enc_Wih1, eWih1, nWhh);   cvt(enc_Whh1, eWhh1, nWhh);
  cvt(dec_Wih0, dWih0, nWih0);  cvt(dec_Whh0, dWhh0, nWhh);
  cvt(dec_Wih1, dWih1, nWhh);   cvt(dec_Whh1, dWhh1, nWhh);
  cvt(dec_linW, linW,  nLin);

  // ---- zero-init state ----
  k_zero_f32 <<<(BATCH*HDIM)/T256, T256, 0, stream>>>(c0, BATCH*HDIM);
  k_zero_f32 <<<(BATCH*HDIM)/T256, T256, 0, stream>>>(c1, BATCH*HDIM);
  k_zero_bf16<<<(BATCH*HDIM)/T256, T256, 0, stream>>>(h0, BATCH*HDIM);
  k_zero_bf16<<<(BATCH*HDIM)/T256, T256, 0, stream>>>(h1, BATCH*HDIM);
  k_zero_i32 <<<1, 64, 0, stream>>>(dei, BATCH);
  k_zero_f32 <<<1, 1, 0, stream>>>(loss, 1);

  const int      GEMM_GRID_4H  = H4 / 64;      // 64 blocks x 4 waves
  const int      GEMM_GRID_LIN = HDIM / 64;    // 16 blocks
  const unsigned GEMM_SHMEM    = 2u * BUF_ELEMS * sizeof(__bf16);  // 34816 B
  const int      EMB_GRID      = (BATCH * EDIM) / T256;
  const int      UPD_GRID      = (BATCH * HDIM) / T256;

  // ---- encoder: 128 sequential timesteps ----
  for (int t = 0; t < SRCLEN; ++t) {
    k_embed<<<EMB_GRID, T256, 0, stream>>>(input_ids, SRCLEN, t, enc_embed, EDIM, xe);
    k_gemm_wmma<<<GEMM_GRID_4H, GEMM_THREADS, GEMM_SHMEM, stream>>>(
        xe, eWih0, EDIM, h0, eWhh0, HDIM, z, H4);
    k_lstm_update<<<UPD_GRID, T256, 0, stream>>>(z, enc_b0, c0, h0);
    k_gemm_wmma<<<GEMM_GRID_4H, GEMM_THREADS, GEMM_SHMEM, stream>>>(
        h0, eWih1, HDIM, h1, eWhh1, HDIM, z, H4);
    k_lstm_update<<<UPD_GRID, T256, 0, stream>>>(z, enc_b1, c1, h1);
  }

  // ---- decoder: 128 sequential timesteps with greedy feedback ----
  for (int t = 0; t < TGTLEN; ++t) {
    k_embed<<<EMB_GRID, T256, 0, stream>>>(dei, 1, 0, dec_embed, EDIM, xe);
    k_gemm_wmma<<<GEMM_GRID_4H, GEMM_THREADS, GEMM_SHMEM, stream>>>(
        xe, dWih0, EDIM, h0, dWhh0, HDIM, z, H4);
    k_lstm_update<<<UPD_GRID, T256, 0, stream>>>(z, dec_b0, c0, h0);
    k_gemm_wmma<<<GEMM_GRID_4H, GEMM_THREADS, GEMM_SHMEM, stream>>>(
        h0, dWih1, HDIM, h1, dWhh1, HDIM, z, H4);
    k_lstm_update<<<UPD_GRID, T256, 0, stream>>>(z, dec_b1, c1, h1);
    k_gemm_wmma<<<GEMM_GRID_LIN, GEMM_THREADS, GEMM_SHMEM, stream>>>(
        h1, linW, HDIM, (const __bf16*)nullptr, (const __bf16*)nullptr, 0, zlin, HDIM);
    k_dec_out<<<BATCH, T256, 0, stream>>>(zlin, dec_linb, tag_ids, t,
                                          out + (long)t * BATCH * HDIM, nll, dei);
    k_step_loss<<<1, 64, 0, stream>>>(nll, tag_ids, t, loss);
  }
}
